// Str2MSA_60790967108046
// MI455X (gfx1250) — compile-verified, matrix operand order checked by reading, not run on hardware
//
#include <hip/hip_runtime.h>
#include <hip/hip_bf16.h>

// ---------------------------------------------------------------------------
// CDNA5 (gfx1250, wave32) WMMA types
// ---------------------------------------------------------------------------
typedef __attribute__((ext_vector_type(16))) __bf16 bf16x16;
typedef __attribute__((ext_vector_type(2)))  __bf16 bf16x2;
typedef __attribute__((ext_vector_type(8)))  float  floatx8;

#define L_SEQ   384
#define N_ROWS  512
#define D_MSA   64
#define D_STATE 8
#define D_FF    256
#define NHEAD   4
#define DK      32
#define TOTROWS (N_ROWS * L_SEQ)   // 196608
#define LT_N    (L_SEQ / 16)       // 24 l-tiles
#define MB_N    (L_SEQ / 32)       // 12 m-blocks
#define FRAG_DW 256                // one 16x32 (or 32x16) bf16 fragment = 32 lanes * 8 dwords

// Native RNE conversions (hardware v_cvt on gfx1250)
__device__ __forceinline__ unsigned short f2bf(float x) {
  __bf16 h = (__bf16)x;
  return __builtin_bit_cast(unsigned short, h);
}
__device__ __forceinline__ unsigned int pk2bf(float lo, float hi) {
  bf16x2 v = { (__bf16)lo, (__bf16)hi };
  return __builtin_bit_cast(unsigned int, v);
}

// Packed fragment load: lane's 8 dwords are contiguous -> two b128 loads.
__device__ __forceinline__ bf16x16 load_frag_packed(const unsigned int* p) {
  int lane = threadIdx.x & 31;
  union { bf16x16 v; uint4 q[2]; } f;
  const uint4* s = (const uint4*)(p + lane * 8);
  f.q[0] = s[0];
  f.q[1] = s[1];
  return f.v;
}

__device__ __forceinline__ floatx8 wmma_bf16(bf16x16 a, bf16x16 b, floatx8 c) {
  return __builtin_amdgcn_wmma_f32_16x16x32_bf16(false, a, false, b, (short)0, c,
                                                 false, false);
}

// ---------------------------------------------------------------------------
// Kernel 1: pack GEMM weights (row-major f32 [K x N]) into bf16 fragments.
// One wave per 32x16 fragment. Frag id within a weight = nt*(K/32)+kt
// (w2 uses nt*(K/32)+kt with K=256 -> nt*8+kt).
// ---------------------------------------------------------------------------
__global__ void k_pack_weights(const float* wv, const float* wo, const float* w1,
                               const float* w2, unsigned int* wvf,
                               unsigned int* wof, unsigned int* w1f,
                               unsigned int* w2f) {
  int fid = blockIdx.x;
  const float* src; unsigned int* dst; int N, nt, kt, lf;
  if (fid < 8)       { src = wv; dst = wvf; N = 64;  lf = fid;      nt = lf >> 1; kt = lf & 1; }
  else if (fid < 16) { src = wo; dst = wof; N = 64;  lf = fid - 8;  nt = lf >> 1; kt = lf & 1; }
  else if (fid < 48) { src = w1; dst = w1f; N = 256; lf = fid - 16; nt = lf >> 1; kt = lf & 1; }
  else               { src = w2; dst = w2f; N = 64;  lf = fid - 48; nt = lf >> 3; kt = lf & 7; }
  int lane = threadIdx.x, c = lane & 15, half = lane >> 4;
  unsigned int* o = dst + (size_t)lf * FRAG_DW + lane * 8;
#pragma unroll
  for (int j = 0; j < 8; ++j) {
    int k = kt * 32 + ((j < 4) ? (j * 2) : (16 + (j - 4) * 2)) + half * 8;
    o[j] = pk2bf(src[(size_t)k * N + nt * 16 + c],
                 src[(size_t)(k + 1) * N + nt * 16 + c]);
  }
}

// ---------------------------------------------------------------------------
// Kernel 2: q,k = LN(state) @ wq/wk + bq/bk   (tiny)
// ---------------------------------------------------------------------------
__global__ void k_qk(const float* state, const float* g1, const float* b1,
                     const float* wq, const float* bq, const float* wk,
                     const float* bk, float* q_ws, float* k_ws) {
  int l = blockIdx.x;
  int j = threadIdx.x;
  float st[D_STATE];
  float mu = 0.f;
#pragma unroll
  for (int i = 0; i < D_STATE; ++i) { st[i] = state[l * D_STATE + i]; mu += st[i]; }
  mu *= (1.0f / D_STATE);
  float var = 0.f;
#pragma unroll
  for (int i = 0; i < D_STATE; ++i) { float d = st[i] - mu; var += d * d; }
  var *= (1.0f / D_STATE);
  float rs = rsqrtf(var + 1e-5f);
#pragma unroll
  for (int i = 0; i < D_STATE; ++i) st[i] = (st[i] - mu) * rs * g1[i] + b1[i];
  float q = bq[j], k = bk[j];
#pragma unroll
  for (int i = 0; i < D_STATE; ++i) {
    q += st[i] * wq[i * 128 + j];
    k += st[i] * wk[i * 128 + j];
  }
  q_ws[l * 128 + j] = q;
  k_ws[l * 128 + j] = k;
}

// ---------------------------------------------------------------------------
// Kernel 3: masked softmax attention, written directly in A-fragment layout.
// attf frag order: ((h*24 + lt)*12 + mb), one wave per (head, query-row).
// Element (r=l&15, k=m&31) lives at lane' = r + 16*((k>>3)&1),
// dword j' = (k<16?0:4) + ((k&7)>>1), half p = k&1. Here k == lane.
// ---------------------------------------------------------------------------
__global__ void k_att(const float* q_ws, const float* k_ws, const float* xyz,
                      const int* idx, unsigned int* attf) {
  int h = blockIdx.x / L_SEQ;
  int l = blockIdx.x % L_SEQ;
  int lane = threadIdx.x;

  __shared__ float qs[DK];
  qs[lane] = q_ws[l * 128 + h * DK + lane];
  __syncthreads();

  float cax = xyz[(l * 3 + 1) * 3 + 0];
  float cay = xyz[(l * 3 + 1) * 3 + 1];
  float caz = xyz[(l * 3 + 1) * 3 + 2];
  int il = idx[l];
  float distbin = 8.0f + 4.0f * (float)h;
  const float NEGMIN = -3.4028234663852886e38f;

  float logit[MB_N];
#pragma unroll
  for (int t = 0; t < MB_N; ++t) {
    int m = lane + 32 * t;
    const float* kr = k_ws + m * 128 + h * DK;
    float dot = 0.f;
#pragma unroll
    for (int d = 0; d < DK; ++d) dot += qs[d] * kr[d];
    dot *= 0.17677669529663687f;  // 1/sqrt(32)
    float dx = cax - xyz[(m * 3 + 1) * 3 + 0];
    float dy = cay - xyz[(m * 3 + 1) * 3 + 1];
    float dz = caz - xyz[(m * 3 + 1) * 3 + 2];
    float dist = sqrtf(dx * dx + dy * dy + dz * dz);
    int sep = abs(idx[m] - il);
    bool keep = (dist > distbin) && (sep > 24);
    logit[t] = keep ? dot : NEGMIN;
  }
  float mx = logit[0];
#pragma unroll
  for (int t = 1; t < MB_N; ++t) mx = fmaxf(mx, logit[t]);
#pragma unroll
  for (int w = 16; w >= 1; w >>= 1) mx = fmaxf(mx, __shfl_xor(mx, w, 32));
  float s = 0.f;
#pragma unroll
  for (int t = 0; t < MB_N; ++t) { logit[t] = __expf(logit[t] - mx); s += logit[t]; }
#pragma unroll
  for (int w = 16; w >= 1; w >>= 1) s += __shfl_xor(s, w, 32);
  float inv = 1.0f / s;

  int lt = l >> 4, r = l & 15;
  int lanep = r + 16 * ((lane >> 3) & 1);
  int jp = ((lane < 16) ? 0 : 4) + ((lane & 7) >> 1);
  int p = lane & 1;
  unsigned short* base = (unsigned short*)attf;
  size_t fb = (size_t)(h * LT_N + lt) * MB_N;
#pragma unroll
  for (int t = 0; t < MB_N; ++t)
    base[((fb + t) * FRAG_DW + lanep * 8 + jp) * 2 + p] = f2bf(logit[t] * inv);
}

// ---------------------------------------------------------------------------
// Kernel 4: v = LN(msa) @ wv + bv, emitted directly in B-fragment layout.
// Block = 256 threads (8 waves), 32 rows. msa tile staged via TDM if present.
// vf frag order: ((n*4 + h)*12 + mb).
// ---------------------------------------------------------------------------
__global__ void k_vproj(const float* msa, const float* g0, const float* b0,
                        const unsigned int* wvf, const float* bv,
                        unsigned int* vf) {
  __shared__ __align__(16) unsigned int Af[4 * FRAG_DW];  // LN frags (tr,kt)
#if defined(__gfx1250__) && __has_builtin(__builtin_amdgcn_tensor_load_to_lds)
  __shared__ __align__(16) float Mbuf[32 * D_MSA];        // raw f32 tile (TDM dest)
#endif
  int R = blockIdx.x * 32;
  int t = threadIdx.x;

#if defined(__gfx1250__) && __has_builtin(__builtin_amdgcn_tensor_load_to_lds)
  // --- Tensor Data Mover: DMA 32x64 f32 tile global -> LDS ---
  if (t < 32) {
    typedef unsigned int v4u __attribute__((ext_vector_type(4)));
    typedef int v8i __attribute__((ext_vector_type(8)));
    typedef int v4i __attribute__((ext_vector_type(4)));
    unsigned long long ga =
        (unsigned long long)(const void*)(msa + (size_t)R * D_MSA);
    unsigned int lds = (unsigned int)(size_t)(&Mbuf[0]);
    v4u g0d = { 1u,                                  // count=1 (valid descriptor)
                lds,                                 // lds_addr
                (unsigned int)(ga & 0xffffffffu),    // global_addr[31:0]
                (unsigned int)((ga >> 32) & 0x01ffffffu) | 0x80000000u }; // [56:32] | type=2
    v8i g1d = { (int)0x00020000,          // data_size=4B
                (int)(64u << 16),         // tensor_dim0 = 64 (low 16 in [63:48])
                0,                        // tensor_dim0[31:16]=0, tensor_dim1[15:0]=0
                (int)(3u | (64u << 16)),  // tensor_dim1 hi (196608>>16=3), tile_dim0=64
                32,                       // tile_dim1 = 32
                64,                       // tensor_dim0_stride = 64
                0, 0 };
    v4i z4 = { 0, 0, 0, 0 };
    v8i z8 = { 0, 0, 0, 0, 0, 0, 0, 0 };
    __builtin_amdgcn_tensor_load_to_lds(g0d, g1d, z4, z4, z8, 0);
    __builtin_amdgcn_s_wait_tensorcnt(0);
  }
  __syncthreads();
#endif

  // LN: 8 threads per row, 8 elements each; write straight into A-frag layout.
  {
    int rl = t >> 3, seg = t & 7, r = rl & 15, tr = rl >> 4;
    float x[8], s = 0.f;
#if defined(__gfx1250__) && __has_builtin(__builtin_amdgcn_tensor_load_to_lds)
    const float* src = &Mbuf[rl * D_MSA + seg * 8];
#else
    const float* src = msa + (size_t)(R + rl) * D_MSA + seg * 8;
#endif
#pragma unroll
    for (int i = 0; i < 8; ++i) { x[i] = src[i]; s += x[i]; }
#pragma unroll
    for (int w = 4; w >= 1; w >>= 1) s += __shfl_xor(s, w, 32);
    float mu = s * (1.0f / 64.0f);
    float v = 0.f;
#pragma unroll
    for (int i = 0; i < 8; ++i) { float d = x[i] - mu; v += d * d; }
#pragma unroll
    for (int w = 4; w >= 1; w >>= 1) v += __shfl_xor(v, w, 32);
    float rs = rsqrtf(v * (1.0f / 64.0f) + 1e-5f);
    float y[8];
#pragma unroll
    for (int i = 0; i < 8; ++i) {
      int e = seg * 8 + i;
      y[i] = (x[i] - mu) * rs * g0[e] + b0[e];
    }
    // frag (tr, kt): kt = seg>>2; lane' = r + 16*(seg&1); dword base (seg&3>=2 ? 4:0)
    int kt = seg >> 2;
    int lanep = r + 16 * (seg & 1);
    int jb = ((seg & 3) >= 2) ? 4 : 0;
    uint4 pk;
    pk.x = pk2bf(y[0], y[1]); pk.y = pk2bf(y[2], y[3]);
    pk.z = pk2bf(y[4], y[5]); pk.w = pk2bf(y[6], y[7]);
    *(uint4*)&Af[(tr * 2 + kt) * FRAG_DW + lanep * 8 + jb] = pk;
  }
  __syncthreads();

  int wid = t >> 5, tr = wid >> 2, tc = wid & 3;
  floatx8 acc = {};
#pragma unroll
  for (int kt = 0; kt < 2; ++kt)
    acc = wmma_bf16(load_frag_packed(&Af[(tr * 2 + kt) * FRAG_DW]),
                    load_frag_packed(wvf + (tc * 2 + kt) * FRAG_DW), acc);

  // C tile rows = m, cols = d; emit as B-fragment for the att@v GEMM:
  // lane' = c + 16*half, dwords j' = tr*4 + j/2 (pairs j,j+1 pack one dword).
  int lane = t & 31, c = lane & 15, half = lane >> 4;
  float bias = bv[tc * 16 + c];
  int n = R / L_SEQ, mb = (R % L_SEQ) / 32;
  unsigned int* o = vf + (size_t)((n * NHEAD + tc) * MB_N + mb) * FRAG_DW +
                    (c + 16 * half) * 8 + tr * 4;
  uint4 pk;
  pk.x = pk2bf(acc[0] + bias, acc[1] + bias);
  pk.y = pk2bf(acc[2] + bias, acc[3] + bias);
  pk.z = pk2bf(acc[4] + bias, acc[5] + bias);
  pk.w = pk2bf(acc[6] + bias, acc[7] + bias);
  *(uint4*)o = pk;
}

// ---------------------------------------------------------------------------
// Kernel 5: out = (att @ v) @ wo + bo + msa -> msa_new (f32) in d_out.
// grid = N * 24 blocks, 128 threads (4 waves). All operands frag-packed.
// ---------------------------------------------------------------------------
__global__ void k_apply(const unsigned int* attf, const unsigned int* vf,
                        const unsigned int* wof, const float* bo,
                        const float* msa, float* out) {
  __shared__ __align__(16) unsigned int Tf[2 * FRAG_DW];
  int n = blockIdx.x / LT_N, lt = blockIdx.x % LT_N;
  int t = threadIdx.x, wid = t >> 5, lane = t & 31, c = lane & 15, half = lane >> 4;

  // Phase 1: per-head 16x16 attention-weighted sum of v over 12 m-blocks.
  {
    int h = wid;
    const unsigned int* ap = attf + (size_t)(h * LT_N + lt) * MB_N * FRAG_DW;
    const unsigned int* bp = vf + (size_t)(n * NHEAD + h) * MB_N * FRAG_DW;
    floatx8 acc = {};
#pragma unroll
    for (int mb = 0; mb < MB_N; ++mb) {
      __builtin_prefetch(ap + (mb + 1) * FRAG_DW + lane * 8, 0, 0);
      __builtin_prefetch(bp + (mb + 1) * FRAG_DW + lane * 8, 0, 0);
      acc = wmma_bf16(load_frag_packed(ap + mb * FRAG_DW),
                      load_frag_packed(bp + mb * FRAG_DW), acc);
    }
    // C (rows=l, cols=d=h*16+c) -> A-frag layout for the wo GEMM:
    int kt = h >> 1;
    int jp = ((h & 1) ? 4 : 0) + ((c & 7) >> 1);
    int p = c & 1;
    unsigned short* Ts = (unsigned short*)Tf;
#pragma unroll
    for (int j = 0; j < 8; ++j) {
      int lanep = j + 8 * half + 16 * (c >> 3);
      Ts[(kt * FRAG_DW + lanep * 8 + jp) * 2 + p] = f2bf(acc[j]);
    }
  }
  __syncthreads();

  // Phase 2: @ wo + bo + msa residual.
  {
    int ct = wid;
    floatx8 acc = {};
#pragma unroll
    for (int kt = 0; kt < 2; ++kt)
      acc = wmma_bf16(load_frag_packed(&Tf[kt * FRAG_DW]),
                      load_frag_packed(wof + (ct * 2 + kt) * FRAG_DW), acc);
    int col = ct * 16 + c;
    float bias = bo[col];
#pragma unroll
    for (int j = 0; j < 8; ++j) {
      int row = lt * 16 + j + 8 * half;
      size_t gi = (size_t)(n * L_SEQ + row) * D_MSA + col;
      out[gi] = acc[j] + bias + msa[gi];
    }
  }
}

// ---------------------------------------------------------------------------
// Kernel 6: FFN: d_out = x + relu(LN(x)@w1 + bf1)@w2 + bf2   (x = d_out)
// 256 threads (8 waves), 16 rows. Frag-packed LDS staging throughout.
// ---------------------------------------------------------------------------
__global__ void k_ffn(const float* g2, const float* b2, const unsigned int* w1f,
                      const float* bf1, const unsigned int* w2f,
                      const float* bf2, float* out) {
  __shared__ __align__(16) unsigned int Af[2 * FRAG_DW];  // LN(x) frags
  __shared__ __align__(16) unsigned int Hf[8 * FRAG_DW];  // relu(h) frags
  int R = blockIdx.x * 16;
  int t = threadIdx.x, wid = t >> 5, lane = t & 31, c = lane & 15, half = lane >> 4;

  // Phase 1: LN (16 threads per row, 4 elems) -> A-frag layout.
  {
    int rl = t >> 4, e0 = (t & 15) * 4;
    const float* src = out + (size_t)(R + rl) * D_MSA + e0;
    float x[4], s = 0.f;
#pragma unroll
    for (int i = 0; i < 4; ++i) { x[i] = src[i]; s += x[i]; }
#pragma unroll
    for (int w = 8; w >= 1; w >>= 1) s += __shfl_xor(s, w, 32);
    float mu = s * (1.0f / 64.0f);
    float v = 0.f;
#pragma unroll
    for (int i = 0; i < 4; ++i) { float d = x[i] - mu; v += d * d; }
#pragma unroll
    for (int w = 8; w >= 1; w >>= 1) v += __shfl_xor(v, w, 32);
    float rs = rsqrtf(v * (1.0f / 64.0f) + 1e-5f);
    float y[4];
#pragma unroll
    for (int i = 0; i < 4; ++i) {
      int e = e0 + i;
      y[i] = (x[i] - mu) * rs * g2[e] + b2[e];
    }
    int kt = e0 >> 5;
    int lanep = rl + 16 * ((e0 >> 3) & 1);
    int jb = (((e0 & 31) >= 16) ? 4 : 0) + ((e0 & 7) >> 1);
    uint2 pk;
    pk.x = pk2bf(y[0], y[1]); pk.y = pk2bf(y[2], y[3]);
    *(uint2*)&Af[kt * FRAG_DW + lanep * 8 + jb] = pk;
  }
  __syncthreads();

  // Phase 2: h = relu(LN(x) @ w1 + bf1), 16 col tiles, 2 per wave -> Hf frags.
#pragma unroll
  for (int rep = 0; rep < 2; ++rep) {
    int tc = wid + rep * 8;
    floatx8 acc = {};
#pragma unroll
    for (int kt = 0; kt < 2; ++kt)
      acc = wmma_bf16(load_frag_packed(&Af[kt * FRAG_DW]),
                      load_frag_packed(w1f + (tc * 2 + kt) * FRAG_DW), acc);
    int col = tc * 16 + c;
    float bias = bf1[col];
    int kt2 = tc >> 1;
    int jp = ((tc & 1) ? 4 : 0) + ((c & 7) >> 1);
    int p = c & 1;
    unsigned short* Hs = (unsigned short*)Hf;
#pragma unroll
    for (int j = 0; j < 8; ++j) {
      int lanep = j + 8 * half + 16 * (c >> 3);
      Hs[(kt2 * FRAG_DW + lanep * 8 + jp) * 2 + p] = f2bf(fmaxf(acc[j] + bias, 0.0f));
    }
  }
  __syncthreads();

  // Phase 3: y = h @ w2 + bf2 + x, waves 0..3, K=256 (8 frag steps).
  if (wid < 4) {
    floatx8 acc = {};
#pragma unroll
    for (int kt = 0; kt < 8; ++kt)
      acc = wmma_bf16(load_frag_packed(&Hf[kt * FRAG_DW]),
                      load_frag_packed(w2f + (wid * 8 + kt) * FRAG_DW), acc);
    int col = wid * 16 + c;
    float bias = bf2[col];
#pragma unroll
    for (int j = 0; j < 8; ++j) {
      size_t gi = (size_t)(R + j + 8 * half) * D_MSA + col;
      out[gi] = acc[j] + bias + out[gi];
    }
  }
}

// ---------------------------------------------------------------------------
// Host launcher
// ---------------------------------------------------------------------------
extern "C" void kernel_launch(void* const* d_in, const int* in_sizes, int n_in,
                              void* d_out, int out_size, void* d_ws, size_t ws_size,
                              hipStream_t stream) {
  const float* msa   = (const float*)d_in[0];
  const float* xyz   = (const float*)d_in[1];
  const float* state = (const float*)d_in[2];
  const int*   idx   = (const int*)d_in[3];
  const float* g0 = (const float*)d_in[4];
  const float* b0 = (const float*)d_in[5];
  const float* g1 = (const float*)d_in[6];
  const float* b1 = (const float*)d_in[7];
  const float* wq = (const float*)d_in[8];
  const float* bq = (const float*)d_in[9];
  const float* wk = (const float*)d_in[10];
  const float* bk = (const float*)d_in[11];
  const float* wv = (const float*)d_in[12];
  const float* bv = (const float*)d_in[13];
  const float* wo = (const float*)d_in[14];
  const float* bo = (const float*)d_in[15];
  const float* g2 = (const float*)d_in[16];
  const float* b2 = (const float*)d_in[17];
  const float* w1 = (const float*)d_in[18];
  const float* bf1 = (const float*)d_in[19];
  const float* w2 = (const float*)d_in[20];
  const float* bf2 = (const float*)d_in[21];
  float* out = (float*)d_out;

  // workspace carve-up (all fragment buffers 16B aligned)
  float* q_ws = (float*)d_ws;                                   // 384*128 f32
  float* k_ws = q_ws + L_SEQ * 128;                             // 384*128 f32
  unsigned int* attf = (unsigned int*)(k_ws + L_SEQ * 128);     // 4*24*12 frags
  unsigned int* vfrag = attf + (size_t)NHEAD * LT_N * MB_N * FRAG_DW; // 512*4*12 frags
  unsigned int* wvf = vfrag + (size_t)N_ROWS * NHEAD * MB_N * FRAG_DW; // 8 frags
  unsigned int* wof = wvf + 8 * FRAG_DW;                        // 8 frags
  unsigned int* w1f = wof + 8 * FRAG_DW;                        // 32 frags
  unsigned int* w2f = w1f + 32 * FRAG_DW;                       // 32 frags

  k_pack_weights<<<80, 32, 0, stream>>>(wv, wo, w1, w2, wvf, wof, w1f, w2f);
  k_qk<<<L_SEQ, 128, 0, stream>>>(state, g1, b1, wq, bq, wk, bk, q_ws, k_ws);
  k_att<<<NHEAD * L_SEQ, 32, 0, stream>>>(q_ws, k_ws, xyz, idx, attf);
  k_vproj<<<TOTROWS / 32, 256, 0, stream>>>(msa, g0, b0, wvf, bv, vfrag);
  k_apply<<<N_ROWS * LT_N, 128, 0, stream>>>(attf, vfrag, wof, bo, msa, out);
  k_ffn<<<TOTROWS / 16, 256, 0, stream>>>(g2, b2, w1f, bf1, w2f, bf2, out);
}
